// LSHSelfAttention_84000970375375
// MI455X (gfx1250) — compile-verified, matrix operand order checked by reading, not run on hardware
//
#include <hip/hip_runtime.h>
#include <stdint.h>

// ---------- problem constants ----------
#define D_MODEL 1024
#define N_HEADS 16
#define D_K     64
#define SEQ     2048
#define BATCH   2
#define MROWS   (BATCH * SEQ)          // 4096 rows of x / ctx
#define QKV_ELEMS ((size_t)BATCH * N_HEADS * SEQ * D_K)   // 4,194,304 per tensor

typedef __attribute__((ext_vector_type(16))) __bf16 v16bf;
typedef __attribute__((ext_vector_type(8)))  float  v8f;

// ---------- WMMA wrapper ----------
__device__ inline v8f wmma_bf16(v16bf a, v16bf b, v8f c) {
    // D = A(16x32 bf16) * B(32x16 bf16) + C(16x16 f32)
    return __builtin_amdgcn_wmma_f32_16x16x32_bf16(false, a, false, b,
                                                   (short)0, c, false, false);
}

// ---------- async copy: one 16B global->LDS transfer per lane ----------
__device__ inline void async_copy_16B(const __bf16* gptr, __bf16* lptr) {
    const uint32_t lds_off = (uint32_t)(uintptr_t)lptr;   // low 32b = LDS offset
    asm volatile("global_load_async_to_lds_b128 %0, %1, off"
                 :: "v"(lds_off), "v"(gptr) : "memory");
}
__device__ inline void wait_async0() {
    asm volatile("s_wait_asynccnt 0" ::: "memory");
}
__device__ inline void wait_ds0() {
    asm volatile("s_wait_dscnt 0" ::: "memory");
}

// ---------- fragment loaders ----------
// Contiguous-K fragment: lane holds a row with K contiguous; element j maps to
// K = ((j>>3)*2 + laneHi)*8 + (j&7) -> two 16B vector loads (global or LDS).
__device__ inline v16bf load_frag_p(const __bf16* p, int laneHi) {
    union { uint4 u[2]; v16bf v; } t;
    t.u[0] = *reinterpret_cast<const uint4*>(p + 8 * laneHi);        // K group 0/1
    t.u[1] = *reinterpret_cast<const uint4*>(p + 16 + 8 * laneHi);   // K group 2/3
    return t.v;
}
__device__ inline v16bf load_frag_k(const __bf16* __restrict__ A, int ld,
                                    int row, int k0, int laneHi) {
    return load_frag_p(A + (size_t)row * ld + k0, laneHi);
}

// ---------- 1a) fp32 -> bf16 linear cast (for x) ----------
__global__ void cast_f32_to_bf16(const float* __restrict__ src,
                                 __bf16* __restrict__ dst, int n) {
    int i = blockIdx.x * blockDim.x + threadIdx.x;
    if (i < n) dst[i] = (__bf16)src[i];
}

// ---------- 1b) fp32 -> bf16 transpose-cast (4 weight matrices) ----------
// in : W  [K][N] fp32 row-major;  out: WT [N][K] bf16, WT[n][k] = W[k][n]
__global__ __launch_bounds__(256)
void transpose_cast_kernel(const float* __restrict__ w0,
                           const float* __restrict__ w1,
                           const float* __restrict__ w2,
                           const float* __restrict__ w3,
                           __bf16* __restrict__ wtAll) {
    __shared__ float tile[32][33];
    const int z = blockIdx.z;
    const float* W = (z == 0) ? w0 : (z == 1) ? w1 : (z == 2) ? w2 : w3;
    __bf16* WT = wtAll + (size_t)z * D_MODEL * D_MODEL;

    const int tx = threadIdx.x, ty = threadIdx.y;      // 32 x 8
    const int n0 = blockIdx.x * 32, k0 = blockIdx.y * 32;
#pragma unroll
    for (int i = 0; i < 4; ++i)
        tile[ty + 8 * i][tx] = W[(size_t)(k0 + ty + 8 * i) * D_MODEL + n0 + tx];
    __syncthreads();
#pragma unroll
    for (int i = 0; i < 4; ++i)
        WT[(size_t)(n0 + ty + 8 * i) * D_MODEL + k0 + tx] =
            (__bf16)tile[tx][ty + 8 * i];
}

// ---------- 2) fused Q/K/V projection: Y = X @ W + b ----------
// A-tile (64x32, shared by all 8 waves of the block) staged via async->LDS,
// double-buffered; B-frags direct b128 from transposed weights.
// Q,K stored bf16 [B,H,S,dk]; V stored TRANSPOSED bf16 [B,H,dk,S].
__global__ __launch_bounds__(256)
void qkv_gemm_kernel(const __bf16* __restrict__ X,      // [4096,1024] bf16
                     const __bf16* __restrict__ WTall,  // [3][N][K] bf16
                     const float* __restrict__ bq,
                     const float* __restrict__ bk,
                     const float* __restrict__ bv,
                     __bf16* __restrict__ qkv) {
    __shared__ alignas(16) __bf16 atile[2][64][32];     // 2 x 4KB

    const int z = blockIdx.z;
    const __bf16* WT = WTall + (size_t)z * D_MODEL * D_MODEL;
    const float* bias = (z == 0) ? bq : (z == 1) ? bk : bv;
    __bf16* out = qkv + (size_t)z * QKV_ELEMS;

    const int t = threadIdx.x;
    const int lane = t & 31, laneL = lane & 15, laneHi = lane >> 4;
    const int wid = blockIdx.x * 8 + (t >> 5);            // 1024 waves
    const int m0 = (wid >> 4) * 64;                       // same for all 8 waves
    const int n0 = (wid & 15) * 64;

    const int arow = t >> 2, ach = (t & 3) * 8;           // 256 x 16B = 4KB tile
    // prologue: stage tile k0=0 into buffer 0
    async_copy_16B(X + (size_t)(m0 + arow) * D_MODEL + ach, &atile[0][arow][ach]);

    v8f acc[4][4] = {};
    for (int k0 = 0, it = 0; k0 < D_MODEL; k0 += 32, ++it) {
        const int buf = it & 1;
        wait_async0();
        __syncthreads();                                  // tile `it` landed
        v16bf a[4], bm[4];
#pragma unroll
        for (int mi = 0; mi < 4; ++mi)
            a[mi] = load_frag_p(&atile[buf][mi * 16 + laneL][0], laneHi);
#pragma unroll
        for (int ni = 0; ni < 4; ++ni)
            bm[ni] = load_frag_k(WT, D_MODEL, n0 + ni * 16 + laneL, k0, laneHi);
        wait_ds0();
        __syncthreads();                                  // all reads retired
        if (k0 + 32 < D_MODEL)                            // overlap next stage
            async_copy_16B(X + (size_t)(m0 + arow) * D_MODEL + k0 + 32 + ach,
                           &atile[buf ^ 1][arow][ach]);
#pragma unroll
        for (int mi = 0; mi < 4; ++mi)
#pragma unroll
            for (int ni = 0; ni < 4; ++ni)
                acc[mi][ni] = wmma_bf16(a[mi], bm[ni], acc[mi][ni]);
    }
    // epilogue: bias + layout-transform store
#pragma unroll
    for (int mi = 0; mi < 4; ++mi) {
#pragma unroll
        for (int ni = 0; ni < 4; ++ni) {
            const int col = n0 + ni * 16 + laneL;
            const float bv_ = bias[col];
            const int h = col >> 6, d = col & 63;
#pragma unroll
            for (int r = 0; r < 8; ++r) {
                const int row = m0 + mi * 16 + r + 8 * laneHi;  // = b*2048 + s
                const int b = row >> 11, s = row & 2047;
                const __bf16 y = (__bf16)(acc[mi][ni][r] + bv_);
                if (z == 2)  // V transposed: [B,H,dk,S]
                    out[((size_t)(b * N_HEADS + h) * D_K + d) * SEQ + s] = y;
                else         // Q,K: [B,H,S,dk]
                    out[((size_t)(b * N_HEADS + h) * SEQ + s) * D_K + d] = y;
            }
        }
    }
}

// ---------- 3) flash attention ----------
// All 8 waves of a block share one (b,h): K/V tiles staged via async->LDS,
// double-buffered; each wave owns one 16-row Q tile.
__global__ __launch_bounds__(256)
void attn_kernel(const __bf16* __restrict__ qkv,   // Q,K:[B,H,S,dk]  V:[B,H,dk,S]
                 __bf16* __restrict__ ctx) {       // [B,S,D] bf16
    __shared__ alignas(16) __bf16 ktile[2][32][64]; // 2 x 4KB
    __shared__ alignas(16) __bf16 vtile[2][64][32]; // 2 x 4KB
    __shared__ alignas(16) __bf16 pbuf[8][16][32];  // per-wave P relayout, 8KB

    const int t = threadIdx.x;
    const int w = t >> 5;
    const int lane = t & 31, laneL = lane & 15, laneHi = lane >> 4;
    const int wg = blockIdx.x * 8 + w;             // 4096 waves total
    const int bh = wg >> 7;                        // same for whole block
    const int qt = wg & 127;
    const int b = bh >> 4, h = bh & 15;

    const __bf16* Qh = qkv + 0 * QKV_ELEMS + (size_t)bh * SEQ * D_K;
    const __bf16* Kh = qkv + 1 * QKV_ELEMS + (size_t)bh * SEQ * D_K;
    const __bf16* Vt = qkv + 2 * QKV_ELEMS + (size_t)bh * D_K * SEQ;  // [dk][S]

    // K tile: 32 rows x 64 cols -> thread t copies row t>>3, 16B chunk t&7
    const int krow = t >> 3, kch = (t & 7) * 8;
    // V tile: 64 rows x 32 cols -> thread t copies row t>>2, 16B chunk t&3
    const int vrow = t >> 2, vch = (t & 3) * 8;

    // prologue: stage kv tile 0
    async_copy_16B(Kh + (size_t)krow * D_K + kch, &ktile[0][krow][kch]);
    async_copy_16B(Vt + (size_t)vrow * SEQ + vch, &vtile[0][vrow][vch]);

    v16bf aq0 = load_frag_k(Qh, D_K, qt * 16 + laneL, 0,  laneHi);
    v16bf aq1 = load_frag_k(Qh, D_K, qt * 16 + laneL, 32, laneHi);

    float mrow[8], lrow[8];
    v8f acc[4] = {};
#pragma unroll
    for (int r = 0; r < 8; ++r) { mrow[r] = -3.0e38f; lrow[r] = 0.f; }

    for (int kv0 = 0, it = 0; kv0 < SEQ; kv0 += 32, ++it) {
        const int buf = it & 1;
        wait_async0();
        __syncthreads();                           // K/V tile `it` landed

        // fragment reads from LDS (ds_load_b128 pairs)
        v16bf bk00 = load_frag_p(&ktile[buf][laneL][0],       laneHi);
        v16bf bk01 = load_frag_p(&ktile[buf][laneL][32],      laneHi);
        v16bf bk10 = load_frag_p(&ktile[buf][16 + laneL][0],  laneHi);
        v16bf bk11 = load_frag_p(&ktile[buf][16 + laneL][32], laneHi);
        v16bf bv_[4];
#pragma unroll
        for (int c = 0; c < 4; ++c)
            bv_[c] = load_frag_p(&vtile[buf][c * 16 + laneL][0], laneHi);
        wait_ds0();
        __syncthreads();                           // all reads retired
        if (kv0 + 32 < SEQ) {                      // overlap next stage
            async_copy_16B(Kh + (size_t)(kv0 + 32 + krow) * D_K + kch,
                           &ktile[buf ^ 1][krow][kch]);
            async_copy_16B(Vt + (size_t)vrow * SEQ + kv0 + 32 + vch,
                           &vtile[buf ^ 1][vrow][vch]);
        }

        // ---- scores: S(16x32) = Q @ K^T
        v8f s0 = {}, s1 = {};
        s0 = wmma_bf16(aq0, bk00, s0);  s0 = wmma_bf16(aq1, bk01, s0);
        s1 = wmma_bf16(aq0, bk10, s1);  s1 = wmma_bf16(aq1, bk11, s1);

        // ---- online softmax over the 32 new columns
#pragma unroll
        for (int r = 0; r < 8; ++r) {
            float x0 = s0[r] * 0.125f;       // 1/sqrt(64)
            float x1 = s1[r] * 0.125f;
            float mx = fmaxf(x0, x1);
#pragma unroll
            for (int msk = 1; msk < 16; msk <<= 1)
                mx = fmaxf(mx, __shfl_xor(mx, msk, 32));
            const float mn   = fmaxf(mrow[r], mx);
            const float corr = __expf(mrow[r] - mn);
            const float p0 = __expf(x0 - mn);
            const float p1 = __expf(x1 - mn);
            float tsum = p0 + p1;
#pragma unroll
            for (int msk = 1; msk < 16; msk <<= 1)
                tsum += __shfl_xor(tsum, msk, 32);
            lrow[r] = lrow[r] * corr + tsum;
            mrow[r] = mn;
            acc[0][r] *= corr; acc[1][r] *= corr;
            acc[2][r] *= corr; acc[3][r] *= corr;
            pbuf[w][r + 8 * laneHi][laneL]      = (__bf16)p0;
            pbuf[w][r + 8 * laneHi][16 + laneL] = (__bf16)p1;
        }
        wait_ds0();                                // wave-local P relayout

        v16bf pa;
#pragma unroll
        for (int j = 0; j < 8; ++j) {
            pa[j]     = pbuf[w][laneL][8 * laneHi + j];
            pa[8 + j] = pbuf[w][laneL][16 + 8 * laneHi + j];
        }
        // ---- ctx += P(16x32) @ V(32x64)
#pragma unroll
        for (int c = 0; c < 4; ++c)
            acc[c] = wmma_bf16(pa, bv_[c], acc[c]);
    }

    // ---- epilogue: normalize and store ctx bf16 in [B,S,D]
#pragma unroll
    for (int c = 0; c < 4; ++c) {
#pragma unroll
        for (int r = 0; r < 8; ++r) {
            const int s = qt * 16 + r + 8 * laneHi;
            const float v = acc[c][r] / lrow[r];
            ctx[((size_t)(b * SEQ + s)) * D_MODEL + h * D_K + c * 16 + laneL] =
                (__bf16)v;
        }
    }
}

// ---------- 4) output projection: out = ctx @ Wo + bo (fp32 store) ----------
__global__ __launch_bounds__(256)
void outproj_gemm_kernel(const __bf16* __restrict__ A,    // ctx [4096,1024] bf16
                         const __bf16* __restrict__ WT,   // wo^T [N][K] bf16
                         const float* __restrict__ bias,  // bo
                         float* __restrict__ out) {       // [4096,1024] fp32
    __shared__ alignas(16) __bf16 atile[2][64][32];

    const int t = threadIdx.x;
    const int lane = t & 31, laneL = lane & 15, laneHi = lane >> 4;
    const int wid = blockIdx.x * 8 + (t >> 5);
    const int m0 = (wid >> 4) * 64;
    const int n0 = (wid & 15) * 64;

    const int arow = t >> 2, ach = (t & 3) * 8;
    async_copy_16B(A + (size_t)(m0 + arow) * D_MODEL + ach, &atile[0][arow][ach]);

    v8f acc[4][4] = {};
    for (int k0 = 0, it = 0; k0 < D_MODEL; k0 += 32, ++it) {
        const int buf = it & 1;
        wait_async0();
        __syncthreads();
        v16bf a[4], bm[4];
#pragma unroll
        for (int mi = 0; mi < 4; ++mi)
            a[mi] = load_frag_p(&atile[buf][mi * 16 + laneL][0], laneHi);
#pragma unroll
        for (int ni = 0; ni < 4; ++ni)
            bm[ni] = load_frag_k(WT, D_MODEL, n0 + ni * 16 + laneL, k0, laneHi);
        wait_ds0();
        __syncthreads();
        if (k0 + 32 < D_MODEL)
            async_copy_16B(A + (size_t)(m0 + arow) * D_MODEL + k0 + 32 + ach,
                           &atile[buf ^ 1][arow][ach]);
#pragma unroll
        for (int mi = 0; mi < 4; ++mi)
#pragma unroll
            for (int ni = 0; ni < 4; ++ni)
                acc[mi][ni] = wmma_bf16(a[mi], bm[ni], acc[mi][ni]);
    }
#pragma unroll
    for (int mi = 0; mi < 4; ++mi) {
#pragma unroll
        for (int ni = 0; ni < 4; ++ni) {
            const int col = n0 + ni * 16 + laneL;
            const float bv_ = bias[col];
#pragma unroll
            for (int r = 0; r < 8; ++r) {
                const int row = m0 + mi * 16 + r + 8 * laneHi;
                out[(size_t)row * D_MODEL + col] = acc[mi][ni][r] + bv_;
            }
        }
    }
}

// ---------- host launcher ----------
extern "C" void kernel_launch(void* const* d_in, const int* in_sizes, int n_in,
                              void* d_out, int out_size, void* d_ws, size_t ws_size,
                              hipStream_t stream) {
    const float* x  = (const float*)d_in[0];
    const float* wq = (const float*)d_in[1];
    const float* bq = (const float*)d_in[2];
    const float* wk = (const float*)d_in[3];
    const float* bk = (const float*)d_in[4];
    const float* wv = (const float*)d_in[5];
    const float* bv = (const float*)d_in[6];
    const float* wo = (const float*)d_in[7];
    const float* bo = (const float*)d_in[8];
    float* out = (float*)d_out;

    // workspace layout (bytes): xbf | wtbf[4] | qkv[3] | ctx   (~48 MB)
    char* ws = (char*)d_ws;
    const size_t X_ELEMS = (size_t)MROWS * D_MODEL;       // 4,194,304
    const size_t W_ELEMS = (size_t)D_MODEL * D_MODEL;     // 1,048,576
    __bf16* xbf  = (__bf16*)(ws);
    __bf16* wtbf = (__bf16*)(ws + X_ELEMS * 2);
    __bf16* qkvb = (__bf16*)(ws + (X_ELEMS + 4 * W_ELEMS) * 2);
    __bf16* ctxb = (__bf16*)(ws + (X_ELEMS + 4 * W_ELEMS + 3 * QKV_ELEMS) * 2);

    // 1) casts: x linear; weights transpose-cast to [N][K]
    cast_f32_to_bf16<<<(int)((X_ELEMS + 255) / 256), 256, 0, stream>>>(x, xbf, (int)X_ELEMS);
    transpose_cast_kernel<<<dim3(32, 32, 4), dim3(32, 8), 0, stream>>>(
        wq, wk, wv, wo, wtbf);

    // 2) Q/K/V projections (1024 waves per z-slice)
    qkv_gemm_kernel<<<dim3(128, 1, 3), 256, 0, stream>>>(xbf, wtbf, bq, bk, bv, qkvb);

    // 3) attention (4096 waves)
    attn_kernel<<<512, 256, 0, stream>>>(qkvb, ctxb);

    // 4) output projection
    outproj_gemm_kernel<<<128, 256, 0, stream>>>(ctxb, wtbf + 3 * W_ELEMS, bo, out);
}
